// GlobalEmdModel_1941325218383
// MI455X (gfx1250) — matrix operand order repaired and checked by reading, not compiled, not run on hardware
//
#include <hip/hip_runtime.h>

#define MT   32          // points per workgroup tile
#define TPB  256         // 8 waves (wave32)
#define NPTS 400000
#define NSEG 64
#define ENC_NEG_INF 0x007FFFFFu   // encode(-inf)

typedef __bf16 bf16x16 __attribute__((ext_vector_type(16)));
typedef float  f32x8   __attribute__((ext_vector_type(8)));

union Frag { bf16x16 v; uint4 q[2]; };

__device__ __forceinline__ unsigned short f2bf(float x) {
  unsigned u = __float_as_uint(x);
  u += 0x7FFFu + ((u >> 16) & 1u);          // round-to-nearest-even
  return (unsigned short)(u >> 16);
}
// order-preserving float -> u32 (for atomicMax-based segment max)
__device__ __forceinline__ unsigned enc_f32(float x) {
  unsigned u = __float_as_uint(x);
  return (u & 0x80000000u) ? ~u : (u | 0x80000000u);
}
__device__ __forceinline__ float dec_f32(unsigned k) {
  unsigned u = (k & 0x80000000u) ? (k ^ 0x80000000u) : ~k;
  return __uint_as_float(u);
}

// One fused linear layer on a 32-row LDS tile with 2x2 register blocking.
// A: LDS bf16 [MT x SIN(row stride)], weights: global bf16 W^T [NW][K] (rows contiguous).
// Each wave owns all 32 rows x (NW/8) columns; per k-step: 4 frag loads -> 4 WMMAs.
template<int K, int NW, int SIN, int SOUT, bool RELU, bool F32OUT>
__device__ __forceinline__ void layer(const unsigned short* __restrict__ sA,
                                      void* __restrict__ sOutV,
                                      const unsigned short* __restrict__ Wt,
                                      const float* __restrict__ bias,
                                      int tid) {
  const int lane = tid & 31;
  const int wave = tid >> 5;
  const int l15  = lane & 15;
  const int hi   = lane >> 4;
  constexpr int STRIP = NW / 8;            // columns owned by this wave
  unsigned short* sOutH = (unsigned short*)sOutV;
  float*          sOutF = (float*)sOutV;

  const unsigned short* aRow0 = sA + (size_t)l15 * SIN;        // rows 0..15
  const unsigned short* aRow1 = sA + (size_t)(16 + l15) * SIN; // rows 16..31

  auto storeC = [&](f32x8& c, int rBase, int cBase) {
    const int col  = cBase + l15;
    const float bv = bias[col];
    const int rOff = rBase + (hi << 3);
#pragma unroll
    for (int v = 0; v < 8; ++v) {
      float f = c[v] + bv;
      if (RELU) f = fmaxf(f, 0.f);
      if constexpr (F32OUT) sOutF[(rOff + v) * SOUT + col] = f;
      else                  sOutH[(rOff + v) * SOUT + col] = f2bf(f);
    }
  };

#pragma unroll
  for (int tg = 0; tg < STRIP / 32; ++tg) {
    const int colBase = wave * STRIP + tg * 32;
    const unsigned short* w0 = Wt + (size_t)(colBase + l15) * K;       // cols +0..15
    const unsigned short* w1 = Wt + (size_t)(colBase + 16 + l15) * K;  // cols +16..31
    f32x8 c00 = {0.f,0.f,0.f,0.f,0.f,0.f,0.f,0.f};
    f32x8 c01 = {0.f,0.f,0.f,0.f,0.f,0.f,0.f,0.f};
    f32x8 c10 = {0.f,0.f,0.f,0.f,0.f,0.f,0.f,0.f};
    f32x8 c11 = {0.f,0.f,0.f,0.f,0.f,0.f,0.f,0.f};
#pragma unroll
    for (int k0 = 0; k0 < K; k0 += 32) {
      Frag a0, a1, b0, b1;
      // A layout (16-bit 16x32): lanes 0-15 K=0..7 & 16..23, lanes 16-31 K=8..15 & 24..31
      const int ka = k0 + (hi << 3);
      a0.q[0] = *(const uint4*)(aRow0 + ka);
      a0.q[1] = *(const uint4*)(aRow0 + ka + 16);
      a1.q[0] = *(const uint4*)(aRow1 + ka);
      a1.q[1] = *(const uint4*)(aRow1 + ka + 16);
      // B layout (16-bit 32x16): lanes 0-15 K=0..7 & 16..23, lanes 16-31 K=8..15 & 24..31
      const int kb = k0 + (hi << 3);
      b0.q[0] = *(const uint4*)(w0 + kb);
      b0.q[1] = *(const uint4*)(w0 + kb + 16);
      b1.q[0] = *(const uint4*)(w1 + kb);
      b1.q[1] = *(const uint4*)(w1 + kb + 16);
      c00 = __builtin_amdgcn_wmma_f32_16x16x32_bf16(false, a0.v, false, b0.v, (short)0, c00, false, false);
      c01 = __builtin_amdgcn_wmma_f32_16x16x32_bf16(false, a0.v, false, b1.v, (short)0, c01, false, false);
      c10 = __builtin_amdgcn_wmma_f32_16x16x32_bf16(false, a1.v, false, b0.v, (short)0, c10, false, false);
      c11 = __builtin_amdgcn_wmma_f32_16x16x32_bf16(false, a1.v, false, b1.v, (short)0, c11, false, false);
    }
    storeC(c00,  0, colBase);
    storeC(c01,  0, colBase + 16);
    storeC(c10, 16, colBase);
    storeC(c11, 16, colBase + 16);
  }
}

// -------- phase 0: weight convert (transpose + f32->bf16), enc-buffer init --------
__global__ void k_convw(const float* __restrict__ W, unsigned short* __restrict__ Wt,
                        int fin, int fout) {
  int i = blockIdx.x * 256 + threadIdx.x;
  if (i < fin * fout) {
    int n = i / fin, k = i - n * fin;                 // Wt[n][k] = W[k][n]
    Wt[i] = f2bf(W[(size_t)k * fout + n]);
  }
}
__global__ void k_init(unsigned* __restrict__ p, int n) {
  int i = blockIdx.x * 256 + threadIdx.x;
  if (i < n) p[i] = ENC_NEG_INF;
}

// -------- phase 1: mlp1 (128->256->512->512) fused + segment max into enc1 --------
__global__ __launch_bounds__(TPB) void k_phase1(
    const float* __restrict__ pos, const int* __restrict__ segId,
    const unsigned short* __restrict__ Wt1, const float* __restrict__ b1,
    const unsigned short* __restrict__ Wt2, const float* __restrict__ b2,
    const unsigned short* __restrict__ Wt3, const float* __restrict__ b3,
    unsigned* __restrict__ enc1) {
  __shared__ unsigned short sIn[MT * 136];   // 128+8 pad
  __shared__ unsigned short sA1[MT * 264];   // 256+8
  __shared__ unsigned short sA2[MT * 520];   // 512+8
  __shared__ float          sH [MT * 520];
  __shared__ int            sSeg[MT];
  const int tid = threadIdx.x;
  const int rb  = blockIdx.x * MT;

  if (tid < MT) sSeg[tid] = segId[rb + tid];
  for (int i = tid; i < MT * 128; i += TPB) {
    int r = i >> 7, c = i & 127;
    sIn[r * 136 + c] = f2bf(pos[(size_t)(rb + r) * 128 + c]);
  }
  __syncthreads();
  layer<128, 256, 136, 264, true,  false>(sIn, sA1, Wt1, b1, tid);
  __syncthreads();
  layer<256, 512, 264, 520, true,  false>(sA1, sA2, Wt2, b2, tid);
  __syncthreads();
  layer<512, 512, 520, 520, false, true >(sA2, sH,  Wt3, b3, tid);
  __syncthreads();
  // segment max over sorted ids: one atomic per (segment-run, column)
  for (int c = tid; c < 512; c += TPB) {
    int cur = sSeg[0];
    float m = -INFINITY;
    for (int r = 0; r < MT; ++r) {
      int s = sSeg[r];
      float v = sH[r * 520 + c];
      if (s != cur) { atomicMax(&enc1[cur * 512 + c], enc_f32(m)); cur = s; m = v; }
      else          { m = fmaxf(m, v); }
    }
    atomicMax(&enc1[cur * 512 + c], enc_f32(m));
  }
}

// -------- phase 2: concat(pos, g) -> mlp2 (640->512->512) + segment max into enc2 --------
__global__ __launch_bounds__(TPB) void k_phase2(
    const float* __restrict__ pos, const int* __restrict__ segId,
    const unsigned* __restrict__ enc1,
    const unsigned short* __restrict__ Wt4, const float* __restrict__ b4,
    const unsigned short* __restrict__ Wt5, const float* __restrict__ b5,
    unsigned* __restrict__ enc2) {
  __shared__ unsigned short sIn[MT * 648];   // 640+8
  __shared__ unsigned short sA4[MT * 520];
  __shared__ float          sZ [MT * 520];
  __shared__ int            sSeg[MT];
  const int tid = threadIdx.x;
  const int rb  = blockIdx.x * MT;

  if (tid < MT) sSeg[tid] = segId[rb + tid];
  __syncthreads();
  for (int i = tid; i < MT * 640; i += TPB) {
    int r = i / 640, c = i - r * 640;
    float v = (c < 128) ? pos[(size_t)(rb + r) * 128 + c]
                        : dec_f32(enc1[sSeg[r] * 512 + (c - 128)]);
    sIn[r * 648 + c] = f2bf(v);
  }
  __syncthreads();
  layer<640, 512, 648, 520, true,  false>(sIn, sA4, Wt4, b4, tid);
  __syncthreads();
  layer<512, 512, 520, 520, false, true >(sA4, sZ,  Wt5, b5, tid);
  __syncthreads();
  for (int c = tid; c < 512; c += TPB) {
    int cur = sSeg[0];
    float m = -INFINITY;
    for (int r = 0; r < MT; ++r) {
      int s = sSeg[r];
      float v = sZ[r * 520 + c];
      if (s != cur) { atomicMax(&enc2[cur * 512 + c], enc_f32(m)); cur = s; m = v; }
      else          { m = fmaxf(m, v); }
    }
    atomicMax(&enc2[cur * 512 + c], enc_f32(m));
  }
}

// -------- phase 3: decode enc2 -> f32 output [64,512] --------
__global__ void k_decode(const unsigned* __restrict__ e, float* __restrict__ o, int n) {
  int i = blockIdx.x * 256 + threadIdx.x;
  if (i < n) o[i] = dec_f32(e[i]);
}

extern "C" void kernel_launch(void* const* d_in, const int* in_sizes, int n_in,
                              void* d_out, int out_size, void* d_ws, size_t ws_size,
                              hipStream_t stream) {
  const float* pos = (const float*)d_in[0];
  const int*   seg = (const int*)d_in[1];
  const float* W1 = (const float*)d_in[2];  const float* b1 = (const float*)d_in[3];
  const float* W2 = (const float*)d_in[4];  const float* b2 = (const float*)d_in[5];
  const float* W3 = (const float*)d_in[6];  const float* b3 = (const float*)d_in[7];
  const float* W4 = (const float*)d_in[8];  const float* b4 = (const float*)d_in[9];
  const float* W5 = (const float*)d_in[10]; const float* b5 = (const float*)d_in[11];

  char* w = (char*)d_ws;
  unsigned short* Wt1 = (unsigned short*)w; w += (size_t)256 * 128 * 2;
  unsigned short* Wt2 = (unsigned short*)w; w += (size_t)512 * 256 * 2;
  unsigned short* Wt3 = (unsigned short*)w; w += (size_t)512 * 512 * 2;
  unsigned short* Wt4 = (unsigned short*)w; w += (size_t)512 * 640 * 2;
  unsigned short* Wt5 = (unsigned short*)w; w += (size_t)512 * 512 * 2;
  unsigned* enc1 = (unsigned*)w;            w += (size_t)NSEG * 512 * 4;
  unsigned* enc2 = (unsigned*)w;            w += (size_t)NSEG * 512 * 4;

  auto cdiv = [](int a, int b) { return (a + b - 1) / b; };

  k_convw<<<cdiv(128 * 256, 256), 256, 0, stream>>>(W1, Wt1, 128, 256);
  k_convw<<<cdiv(256 * 512, 256), 256, 0, stream>>>(W2, Wt2, 256, 512);
  k_convw<<<cdiv(512 * 512, 256), 256, 0, stream>>>(W3, Wt3, 512, 512);
  k_convw<<<cdiv(640 * 512, 256), 256, 0, stream>>>(W4, Wt4, 640, 512);
  k_convw<<<cdiv(512 * 512, 256), 256, 0, stream>>>(W5, Wt5, 512, 512);
  k_init<<<cdiv(2 * NSEG * 512, 256), 256, 0, stream>>>(enc1, 2 * NSEG * 512); // enc1+enc2 contiguous

  k_phase1<<<NPTS / MT, TPB, 0, stream>>>(pos, seg, Wt1, b1, Wt2, b2, Wt3, b3, enc1);
  k_phase2<<<NPTS / MT, TPB, 0, stream>>>(pos, seg, enc1, Wt4, b4, Wt5, b5, enc2);
  k_decode<<<cdiv(NSEG * 512, 256), 256, 0, stream>>>(enc2, (float*)d_out, NSEG * 512);
}
